// CausalSelfAttention_56169582297269
// MI455X (gfx1250) — matrix operand order repaired and checked by reading, not compiled, not run on hardware
//
#include <hip/hip_runtime.h>
#include <hip/hip_bf16.h>

// ---------------------------------------------------------------------------
// CDNA5 (gfx1250) causal self-attention: bf16 WMMA GEMMs + flash attention.
// wave32; WMMA shape 16x16x32 bf16 -> f32.
// GEMM: 128x128 workgroup tile, 8 waves of 32x64 (2x4 WMMA frags) -> 8 WMMAs
// per 6 fragment loads.  Software-pipelined LDS staging hides global latency
// behind matrix math.
// ---------------------------------------------------------------------------

typedef __attribute__((ext_vector_type(16))) __bf16 bf16x16;
typedef __attribute__((ext_vector_type(8)))  float  f32x8;

union Frag {
  bf16x16 v;
  uint4   u[2];
};

__device__ __forceinline__ f32x8 wmma_bf16(const Frag a, const Frag b, f32x8 c) {
  return __builtin_amdgcn_wmma_f32_16x16x32_bf16(false, a.v, false, b.v,
                                                 (short)0, c, false, false);
}

// A-fragment 16x32 (M x K), source row-major [row][k], ld in elements.
// ISA layout: lanes 0-15 row M=lane, elems 0..7 = K0..7, 8..15 = K16..23;
//             lanes 16-31 same rows, K+8 / K+24.
__device__ __forceinline__ Frag load_frag_a(const __bf16* p, int ld) {
  const int lane = threadIdx.x & 31;
  const int row  = lane & 15;
  const int kb   = (lane >> 4) << 3;       // 0 or 8
  Frag f;
  f.u[0] = *(const uint4*)(p + row * ld + kb);
  f.u[1] = *(const uint4*)(p + row * ld + 16 + kb);
  return f;
}

// B-fragment 32x16 (K x N), source laid out transposed: Bt[col][k] row-major.
// ISA layout: lane col N=lane%16; lanes 0-15 elems = K0..7,K16..23; lanes 16-31 = K8..15,K24..31.
__device__ __forceinline__ Frag load_frag_b(const __bf16* p, int ld) {
  const int lane = threadIdx.x & 31;
  const int row  = lane & 15;
  const int kb   = (lane >> 4) << 3;       // 0 or 8
  const __bf16* q = p + row * ld + kb;
  Frag f;
  f.u[0] = *(const uint4*)(q);
  f.u[1] = *(const uint4*)(q + 16);
  return f;
}

// ---------------------------------------------------------------------------
// Converters
// ---------------------------------------------------------------------------
__global__ void cvt_f32_to_bf16(const float* __restrict__ src,
                                __bf16* __restrict__ dst, int n) {
  int i = (blockIdx.x * blockDim.x + threadIdx.x) * 8;
  if (i + 7 < n) {
    float4 f0 = *(const float4*)(src + i);
    float4 f1 = *(const float4*)(src + i + 4);
    union { __bf16 h[8]; uint4 u; } pk;
    pk.h[0] = (__bf16)f0.x; pk.h[1] = (__bf16)f0.y;
    pk.h[2] = (__bf16)f0.z; pk.h[3] = (__bf16)f0.w;
    pk.h[4] = (__bf16)f1.x; pk.h[5] = (__bf16)f1.y;
    pk.h[6] = (__bf16)f1.z; pk.h[7] = (__bf16)f1.w;
    *(uint4*)(dst + i) = pk.u;
  }
}

// dst[n*K + k] = (bf16) src[k*N + n]   (pre-transpose weights to [N,K])
__global__ void transpose_cvt(const float* __restrict__ src,
                              __bf16* __restrict__ dst, int K, int N) {
  int idx = blockIdx.x * blockDim.x + threadIdx.x;
  if (idx >= K * N) return;
  int n = idx / K, k = idx - n * K;
  dst[idx] = (__bf16)src[k * N + n];
}

// ---------------------------------------------------------------------------
// WMMA GEMM:  out[M,N] = A[M,K](bf16) * Bt[N,K]^T(bf16) + bias
// mode 0: f32 output.  mode 1: scatter into Q[B,H,T,D], K[B,H,T,D], V^T[B,H,D,T]
// Workgroup tile 128x128, 8 waves of 32x64 (2x4 WMMA frags), K step 32.
// ---------------------------------------------------------------------------
#define GM 128
#define GN 128
#define GK 32
#define LDSA 40   // 32 + 8 pad (16B-multiple row stride)
#define LDSB 40

__global__ __launch_bounds__(256, 2)
void gemm_bf16_wmma(const __bf16* __restrict__ A, const __bf16* __restrict__ Bt,
                    const float* __restrict__ bias,
                    float* __restrict__ outF,
                    __bf16* __restrict__ qb, __bf16* __restrict__ kb,
                    __bf16* __restrict__ vTb,
                    int M, int N, int K, int mode) {
  __shared__ __align__(16) __bf16 sA[GM * LDSA];
  __shared__ __align__(16) __bf16 sB[GN * LDSB];

  const int tid   = threadIdx.x;
  const int lane  = tid & 31;
  const int wave  = tid >> 5;
  const int wm    = wave & 3;               // 0..3 -> 32-row strip
  const int wn    = wave >> 2;              // 0..1 -> 64-col strip
  const int mbase = blockIdx.y * GM;
  const int nbase = blockIdx.x * GN;

  f32x8 acc[2][4] = {};

  const int arow = tid >> 2;                // 0..63 (two row batches)
  const int acol = (tid & 3) << 3;          // 0,8,16,24

  const __bf16* pa0 = A  + (size_t)(mbase + arow) * K + acol;
  const __bf16* pa1 = A  + (size_t)(mbase + arow + 64) * K + acol;
  const __bf16* pb0 = Bt + (size_t)(nbase + arow) * K + acol;
  const __bf16* pb1 = Bt + (size_t)(nbase + arow + 64) * K + acol;

  // software pipeline: registers hold tile k+1 while tile k is computed
  uint4 ra0 = *(const uint4*)(pa0);
  uint4 ra1 = *(const uint4*)(pa1);
  uint4 rb0 = *(const uint4*)(pb0);
  uint4 rb1 = *(const uint4*)(pb1);

  for (int k0 = 0; k0 < K; k0 += GK) {
    *(uint4*)(sA + arow * LDSA + acol)        = ra0;
    *(uint4*)(sA + (arow + 64) * LDSA + acol) = ra1;
    *(uint4*)(sB + arow * LDSB + acol)        = rb0;
    *(uint4*)(sB + (arow + 64) * LDSB + acol) = rb1;
    __syncthreads();

    const int kn = k0 + GK;
    if (kn < K) {           // issue next tile's global loads early
      ra0 = *(const uint4*)(pa0 + kn);
      ra1 = *(const uint4*)(pa1 + kn);
      rb0 = *(const uint4*)(pb0 + kn);
      rb1 = *(const uint4*)(pb1 + kn);
      __builtin_prefetch(pa0 + kn + GK, 0, 1);
      __builtin_prefetch(pb0 + kn + GK, 0, 1);
    }

    Frag a0 = load_frag_a(sA + (wm * 32 + 0)  * LDSA, LDSA);
    Frag a1 = load_frag_a(sA + (wm * 32 + 16) * LDSA, LDSA);
#pragma unroll
    for (int j = 0; j < 4; ++j) {
      Frag bj = load_frag_b(sB + (wn * 64 + j * 16) * LDSB, LDSB);
      acc[0][j] = wmma_bf16(a0, bj, acc[0][j]);
      acc[1][j] = wmma_bf16(a1, bj, acc[1][j]);
    }
    __syncthreads();
  }

  const int h8 = (lane >> 4) << 3;          // 0 or 8 (C-frag row half)
  const int cn = lane & 15;                 // C-frag column
#pragma unroll
  for (int i = 0; i < 2; ++i) {
#pragma unroll
    for (int j = 0; j < 4; ++j) {
      const int colb = nbase + wn * 64 + j * 16 + cn;
      const float bv = bias[colb];
#pragma unroll
      for (int r = 0; r < 8; ++r) {
        const int row = mbase + wm * 32 + i * 16 + h8 + r;
        const float val = acc[i][j][r] + bv;
        if (mode == 0) {
          outF[(size_t)row * N + colb] = val;
        } else {
          // N = 3*1024; split into q/k/v and heads (H=16, D=64, T=2048)
          const int sec = colb >> 10;
          const int c   = colb & 1023;
          const int h   = c >> 6;
          const int d   = c & 63;
          const int bb  = row >> 11;
          const int t   = row & 2047;
          const __bf16 hv = (__bf16)val;
          if (sec == 0)
            qb[(((size_t)(bb * 16 + h) * 2048) + t) * 64 + d] = hv;
          else if (sec == 1)
            kb[(((size_t)(bb * 16 + h) * 2048) + t) * 64 + d] = hv;
          else
            vTb[(((size_t)(bb * 16 + h) * 64) + d) * 2048 + t] = hv;
        }
      }
    }
  }
}

// ---------------------------------------------------------------------------
// Flash attention forward (causal).  One WG per (b, h, 128-row q block).
// 8 waves; each wave owns 16 q rows.  K blocks of 32 columns staged in LDS,
// software-pipelined: block jb+1's global loads issue before block jb's math.
// Q[B,H,T,D] bf16, K[B,H,T,D] bf16 (Bt layout for Q*K^T),
// V^T[B,H,D,T] bf16 (Bt layout for P*V).  Output Y[B,T,C] bf16.
// ---------------------------------------------------------------------------
#define SK_LD 72   // 64 + 8 pad
#define SV_LD 40   // 32 + 8 pad
#define SP_LD 40

__global__ __launch_bounds__(256, 2)
void attn_fwd_wmma(const __bf16* __restrict__ Q, const __bf16* __restrict__ Kt,
                   const __bf16* __restrict__ Vt, __bf16* __restrict__ Y) {
  const int qblk = blockIdx.x;   // 0..15 (128 rows each)
  const int h    = blockIdx.y;   // 0..15
  const int b    = blockIdx.z;   // 0..3
  const int tid  = threadIdx.x;
  const int lane = tid & 31;
  const int wave = tid >> 5;

  const __bf16* qh = Q  + (size_t)(b * 16 + h) * 2048 * 64;
  const __bf16* kh = Kt + (size_t)(b * 16 + h) * 2048 * 64;
  const __bf16* vh = Vt + (size_t)(b * 16 + h) * 64 * 2048;

  __shared__ __align__(16) __bf16 sK[32 * SK_LD];
  __shared__ __align__(16) __bf16 sV[64 * SV_LD];
  __shared__ __align__(16) __bf16 sP[8][16 * SP_LD];

  const int qr0 = qblk * 128 + wave * 16;
  const int h8  = (lane >> 4) << 3;
  const int cn  = lane & 15;

  // Q fragments: 16 rows x 64 d -> two 16x32 A-frags, loaded once from global.
  Frag qf[2];
  {
    const int row = qr0 + (lane & 15);
    const int kb8 = (lane >> 4) << 3;
    const __bf16* p = qh + (size_t)row * 64;
    qf[0].u[0] = *(const uint4*)(p + kb8);
    qf[0].u[1] = *(const uint4*)(p + 16 + kb8);
    qf[1].u[0] = *(const uint4*)(p + 32 + kb8);
    qf[1].u[1] = *(const uint4*)(p + 48 + kb8);
  }

  f32x8 o[4] = {};
  float m_i[8], l_i[8];
#pragma unroll
  for (int r = 0; r < 8; ++r) { m_i[r] = -3.0e38f; l_i[r] = 0.0f; }

  const int nblk = (qblk + 1) * 4;   // process j blocks up to the diagonal

  // per-thread staging slices (one uint4 each)
  const int kr = tid >> 3, kc = (tid & 7) << 3;   // K block: 32 rows x 64 cols
  const int vr = tid >> 2, vc = (tid & 3) << 3;   // V block: 64 rows x 32 cols

  uint4 rk = *(const uint4*)(kh + (size_t)kr * 64 + kc);
  uint4 rv = *(const uint4*)(vh + (size_t)vr * 2048 + vc);

  for (int jb = 0; jb < nblk; ++jb) {
    *(uint4*)(sK + kr * SK_LD + kc) = rk;
    *(uint4*)(sV + vr * SV_LD + vc) = rv;
    __syncthreads();

    if (jb + 1 < nblk) {            // overlap next block's loads with math
      const int jn = (jb + 1) * 32;
      rk = *(const uint4*)(kh + (size_t)(jn + kr) * 64 + kc);
      rv = *(const uint4*)(vh + (size_t)vr * 2048 + jn + vc);
    }

    const int j0 = jb * 32;

    // S = (Q * K^T) * scale : 16x32 = two 16x16 tiles, accumulate over d-halves
    f32x8 s0 = {}, s1 = {};
#pragma unroll
    for (int i = 0; i < 2; ++i) {
      s0 = wmma_bf16(qf[i], load_frag_b(sK + i * 32, SK_LD), s0);
      s1 = wmma_bf16(qf[i], load_frag_b(sK + 16 * SK_LD + i * 32, SK_LD), s1);
    }

    // scale + causal mask (finite sentinel; col 0 of block 0 is never masked)
    const float scale = 0.125f;   // 1/sqrt(64)
#pragma unroll
    for (int r = 0; r < 8; ++r) {
      const int rg = qr0 + h8 + r;
      float v0 = s0[r] * scale;
      float v1 = s1[r] * scale;
      if (j0 + cn      > rg) v0 = -3.0e38f;
      if (j0 + 16 + cn > rg) v1 = -3.0e38f;
      s0[r] = v0; s1[r] = v1;
    }

    // online softmax per row (rows live in VGPR index; cols across 16-lane half)
    float p0[8], p1[8], al[8];
#pragma unroll
    for (int r = 0; r < 8; ++r) {
      float t = fmaxf(s0[r], s1[r]);
#pragma unroll
      for (int mk = 1; mk < 16; mk <<= 1) t = fmaxf(t, __shfl_xor(t, mk, 32));
      const float mn = fmaxf(m_i[r], t);
      const float a  = __expf(m_i[r] - mn);
      m_i[r] = mn;
      const float e0 = __expf(s0[r] - mn);
      const float e1 = __expf(s1[r] - mn);
      float rs = e0 + e1;
#pragma unroll
      for (int mk = 1; mk < 16; mk <<= 1) rs += __shfl_xor(rs, mk, 32);
      l_i[r] = l_i[r] * a + rs;
      al[r] = a; p0[r] = e0; p1[r] = e1;
    }
#pragma unroll
    for (int dt = 0; dt < 4; ++dt)
#pragma unroll
      for (int r = 0; r < 8; ++r) o[dt][r] *= al[r];

    // re-layout P: C-frag -> per-wave LDS scratch -> A-frag (LDS is in-order per wave)
    __bf16* sp = &sP[wave][0];
#pragma unroll
    for (int r = 0; r < 8; ++r) {
      sp[(h8 + r) * SP_LD + cn]      = (__bf16)p0[r];
      sp[(h8 + r) * SP_LD + 16 + cn] = (__bf16)p1[r];
    }
    Frag pf = load_frag_a(sp, SP_LD);

    // O += P * V  (4 d-tiles of 16)
#pragma unroll
    for (int dt = 0; dt < 4; ++dt) {
      Frag vf = load_frag_b(sV + dt * 16 * SV_LD, SV_LD);
      o[dt] = wmma_bf16(pf, vf, o[dt]);
    }
    __syncthreads();
  }

  // normalize + store Y[B,T,C] bf16
#pragma unroll
  for (int dt = 0; dt < 4; ++dt) {
#pragma unroll
    for (int r = 0; r < 8; ++r) {
      const int row = qr0 + h8 + r;
      const int col = h * 64 + dt * 16 + cn;
      const float val = o[dt][r] / l_i[r];
      Y[((size_t)(b * 2048 + row)) * 1024 + col] = (__bf16)val;
    }
  }
}

// ---------------------------------------------------------------------------
// Host-side launch
// ---------------------------------------------------------------------------
extern "C" void kernel_launch(void* const* d_in, const int* in_sizes, int n_in,
                              void* d_out, int out_size, void* d_ws, size_t ws_size,
                              hipStream_t stream) {
  (void)in_sizes; (void)n_in; (void)out_size; (void)ws_size;

  const float* x      = (const float*)d_in[0];  // [4,2048,1024]
  const float* w_attn = (const float*)d_in[1];  // [1024,3072]
  const float* b_attn = (const float*)d_in[2];  // [3072]
  const float* w_proj = (const float*)d_in[3];  // [1024,1024]
  const float* b_proj = (const float*)d_in[4];  // [1024]
  float* out = (float*)d_out;                   // [4,2048,1024]

  const int Bb = 4, T = 2048, C = 1024, H = 16;
  const int M = Bb * T;            // 8192
  const int N3 = 3 * C;            // 3072

  char* ws = (char*)d_ws;
  size_t off = 0;
  auto take = [&](size_t bytes) -> char* {
    char* p = ws + off;
    off = (off + bytes + 255) & ~(size_t)255;
    return p;
  };
  __bf16* xh    = (__bf16*)take((size_t)M * C * 2);        // x in bf16
  __bf16* waT   = (__bf16*)take((size_t)N3 * C * 2);       // w_attn^T [3C,C]
  __bf16* wpT   = (__bf16*)take((size_t)C * C * 2);        // w_proj^T [C,C]
  __bf16* qbuf  = (__bf16*)take((size_t)M * C * 2);        // [B,H,T,D]
  __bf16* kbuf  = (__bf16*)take((size_t)M * C * 2);        // [B,H,T,D]
  __bf16* vTbuf = (__bf16*)take((size_t)M * C * 2);        // [B,H,D,T]
  __bf16* yh    = (__bf16*)take((size_t)M * C * 2);        // attention out [B,T,C]

  // 1) convert activations / transpose-convert weights to bf16
  {
    const int n = M * C;
    cvt_f32_to_bf16<<<n / 2048, 256, 0, stream>>>(x, xh, n);
    const int ta = C * N3;
    transpose_cvt<<<(ta + 255) / 256, 256, 0, stream>>>(w_attn, waT, C, N3);
    const int tp = C * C;
    transpose_cvt<<<(tp + 255) / 256, 256, 0, stream>>>(w_proj, wpT, C, C);
  }

  // 2) QKV GEMM with head-split scatter (V stored transposed)
  gemm_bf16_wmma<<<dim3(N3 / GN, M / GM), 256, 0, stream>>>(
      xh, waT, b_attn, nullptr, qbuf, kbuf, vTbuf, M, N3, C, 1);

  // 3) causal flash attention
  attn_fwd_wmma<<<dim3(T / 128, H, Bb), 256, 0, stream>>>(qbuf, kbuf, vTbuf, yh);

  // 4) output projection (f32 out + bias)
  gemm_bf16_wmma<<<dim3(C / GN, M / GM), 256, 0, stream>>>(
      yh, wpT, b_proj, out, nullptr, nullptr, nullptr, M, C, C, 0);
}
